// VideoGenerator_40973988004548
// MI455X (gfx1250) — compile-verified, matrix operand order checked by reading, not run on hardware
//
#include <hip/hip_runtime.h>
#include <cstring>
#include <math.h>

// ---------------------------------------------------------------------------
// CDNA5 (gfx1250) StyleGAN-V video generator.
// Every matmul / conv (implicit GEMM) runs on v_wmma_f32_16x16x32_bf16.
// Weights are pre-packed into the exact bf16 fragment layout (one v16bf load
// per wave per K-step); activations are pre-modulated and stored as bf16 so
// the im2col gather is a branchless u16 load + cndmask. Conv kernels use
// NT=4 N-tile register blocking: one A gather feeds 4 back-to-back WMMAs.
// ---------------------------------------------------------------------------

typedef __attribute__((ext_vector_type(16))) __bf16         v16bf;
typedef __attribute__((ext_vector_type(16))) unsigned short v16u;
typedef __attribute__((ext_vector_type(8)))  float          v8f;

__device__ __forceinline__ int kfrag(int half, int v) {
  // 16-bit A/B fragment striping (ISA 7.12.2): VGPR v<4 -> K = 8*half+2v,
  // v>=4 -> K = 16 + 8*half + 2(v-4); each VGPR packs 2 adjacent K values.
  return (v < 4) ? (half * 8 + 2 * v) : (16 + half * 8 + 2 * (v - 4));
}

struct GemmP {
  const float*  A;       // linear A (M x K) f32, row = r / a_row_div
  const float*  Amul;    // optional elementwise multiplier (styled linear)
  const __bf16* Xs;      // packed modulated activations (bf16, NCHW)
  const __bf16* Bp;      // packed weight fragments (bf16)
  const float*  Bias;    // per-col bias
  const float*  Dmod;    // per-(n, out_ch) demodulation
  const float*  Noise;   // (BT,1,OH,OW)
  const float*  NoiseW;  // scalar noise weight (device)
  float* Y;
  int M, N, K, Kb;       // Kb = K/32 (all K are multiples of 32)
  int a_row_div, amul_row_div;
  int IC, H, W, OH, OW, pad;
  float bias_scale;
  int act;               // 0 none, 1 fused leaky-relu * sqrt(2)
  int out_mode;          // 0: Y[r*N+c]  1: NCHW Y[((n*N+c)*OH+oh)*OW+ow]
};

// ---------------- WMMA implicit-GEMM mega kernel ------------------------
// AMODE: 0 linear, 1 conv (im2col, KSZ, pad), 2 conv-transpose (dilate-2)
// NT: number of adjacent 16-wide N tiles per wave (register blocking)
template<int AMODE, int KSZ, bool AMUL, int NT>
__global__ __launch_bounds__(256) void wmma_gemm(GemmP p) {
  const int lane   = threadIdx.x & 31;
  const int wave   = threadIdx.x >> 5;
  const int tilesN = (p.N + 15) >> 4;
  const int tilesM = (p.M + 15) >> 4;
  const int tilesNg = (tilesN + NT - 1) / NT;
  const int tile   = blockIdx.x * 8 + wave;
  if (tile >= tilesM * tilesNg) return;
  const int tm = tile / tilesNg, tng = tile - tm * tilesNg;
  const int half = lane >> 4;          // K-half selector
  const int lidx = lane & 15;          // row (A) / col (B) within tile
  const int rowA = tm * 16 + lidx;
  const bool rowOK = rowA < p.M;

  // ---- per-lane hoisted row state ----
  int n = 0, oh = 0, ow = 0;
  size_t xbase = 0;
  const float* arow = nullptr;
  const float* amrow = nullptr;
  if (AMODE == 0) {
    int rc = rowOK ? rowA : 0;
    arow = p.A + (size_t)(rc / p.a_row_div) * p.K;
    if (AMUL) amrow = p.Amul + (size_t)(rc / p.amul_row_div) * p.K;
  } else {
    int hw = p.OH * p.OW;
    int rc = rowOK ? rowA : 0;
    n = rc / hw; int rem = rc - n * hw;
    oh = rem / p.OW; ow = rem - oh * p.OW;
    xbase = (size_t)n * p.IC * p.H * p.W;
  }
  const unsigned short* xsu = (const unsigned short*)p.Xs;
  const __bf16* bp[NT];
#pragma unroll
  for (int j = 0; j < NT; ++j) {
    int tc = tng * NT + j; if (tc > tilesN - 1) tc = tilesN - 1;
    bp[j] = p.Bp + ((size_t)tc * p.Kb * 32 + lane) * 16;
  }

  v8f acc[NT];
#pragma unroll
  for (int j = 0; j < NT; ++j) acc[j] = (v8f){};

  for (int kbi = 0; kbi < p.Kb; ++kbi) {
    const int kb = kbi * 32;
    v16bf b[NT];
#pragma unroll
    for (int j = 0; j < NT; ++j)
      b[j] = *(const v16bf*)(bp[j] + (size_t)kbi * 32 * 16);
    v16bf a;
    if (AMODE == 0) {
#pragma unroll
      for (int v = 0; v < 8; ++v) {
        int k0 = kb + kfrag(half, v);
        float x0 = arow[k0], x1 = arow[k0 + 1];
        if (AMUL) { x0 *= amrow[k0]; x1 *= amrow[k0 + 1]; }
        if (!rowOK) { x0 = 0.f; x1 = 0.f; }
        a[2 * v + 0] = (__bf16)x0;
        a[2 * v + 1] = (__bf16)x1;
      }
    } else {
      v16u au;
#pragma unroll
      for (int e = 0; e < 16; ++e) {
        const int v = e >> 1, j = e & 1;
        int k = kb + kfrag(half, v) + j;
        int i = k / (KSZ * KSZ); int t = k - i * (KSZ * KSZ);
        int kh = t / KSZ, kw = t - kh * KSZ;
        bool ok; int ih, iw;
        if (AMODE == 1) {
          ih = oh + kh - p.pad; iw = ow + kw - p.pad;
          ok = ((unsigned)ih < (unsigned)p.H) & ((unsigned)iw < (unsigned)p.W);
        } else {  // conv_transpose: dilated-by-2 input, pad 2 (KSZ==3)
          int u = oh + kh - 2, w2 = ow + kw - 2;
          ok = (u >= 0) & (w2 >= 0) & (((u | w2) & 1) == 0);
          ih = u >> 1; iw = w2 >> 1;
          ok = ok & (ih < p.H) & (iw < p.W);
        }
        size_t idx = xbase + ((size_t)i * p.H + ih) * p.W + iw;
        idx = ok ? idx : (size_t)0;          // clamp -> unconditional load
        unsigned short bits = xsu[idx];
        au[e] = ok ? bits : (unsigned short)0;
      }
      a = __builtin_bit_cast(v16bf, au);
    }
#pragma unroll
    for (int j = 0; j < NT; ++j)
      acc[j] = __builtin_amdgcn_wmma_f32_16x16x32_bf16(
          false, a, false, b[j], (short)0, acc[j], false, false);
  }

  // ---- epilogue: C/D layout -> VGPR v, lanes 0-15: M=v, 16-31: M=8+v ----
#pragma unroll
  for (int j = 0; j < NT; ++j) {
    const int c = (tng * NT + j) * 16 + lidx;
    if (c >= p.N) continue;
#pragma unroll
    for (int v = 0; v < 8; ++v) {
      int r = tm * 16 + half * 8 + v;
      if (r >= p.M) continue;
      float y = acc[j][v];
      int hw = p.OH * p.OW;
      int nn = r, rem = 0;
      if (p.out_mode) { nn = r / hw; rem = r - nn * hw; }
      if (p.Dmod)  y *= p.Dmod[nn * p.N + c];
      if (p.Noise) y += p.NoiseW[0] * p.Noise[(size_t)nn * hw + rem];
      if (p.Bias)  y += p.Bias[c] * p.bias_scale;
      if (p.act)   y = (y > 0.f ? y : 0.2f * y) * 1.4142135623730951f;
      if (p.out_mode)
        p.Y[(((size_t)nn * p.N + c) * p.OH + rem / p.OW) * p.OW + (rem % p.OW)] = y;
      else
        p.Y[(size_t)r * p.N + c] = y;
    }
  }
}

// ---------------- weight packing into fragment layout -------------------
__global__ void pack_w_k(const float* __restrict__ Wsrc, __bf16* __restrict__ Wp,
                         int N, int K, int Kb, float scale, int flip) {
  size_t idx = (size_t)blockIdx.x * blockDim.x + threadIdx.x;
  size_t total = (size_t)((N + 15) >> 4) * Kb * 512;
  if (idx >= total) return;
  int e = idx & 15; size_t q = idx >> 4;
  int lane = (int)(q & 31); q >>= 5;
  int kb = (int)(q % Kb); int tn = (int)(q / Kb);
  int half = lane >> 4, lidx = lane & 15;
  int v = e >> 1, j = e & 1;
  int k = kb * 32 + ((v < 4) ? (half * 8 + 2 * v) : (16 + half * 8 + 2 * (v - 4))) + j;
  int col = tn * 16 + lidx;
  float val = 0.f;
  if (col < N && k < K) {
    int kk = k;
    if (flip) {  // spatially flipped 3x3 kernel for conv_transpose
      int i = k / 9, t = k - i * 9;
      int kh = t / 3, kw = t - kh * 3;
      kk = i * 9 + (2 - kh) * 3 + (2 - kw);
    }
    val = Wsrc[(size_t)col * K + kk] * scale;
  }
  Wp[idx] = (__bf16)val;
}

// ---------------- activation modulate + bf16 pack -----------------------
__global__ void pack_xs_k(const float* __restrict__ X, const float* __restrict__ S,
                          __bf16* __restrict__ xs, int BT, int IC, int HW,
                          int x_n_stride) {
  size_t idx = (size_t)blockIdx.x * blockDim.x + threadIdx.x;
  size_t total = (size_t)BT * IC * HW;
  if (idx >= total) return;
  int hw = idx % HW; size_t q = idx / HW;
  int i = (int)(q % IC); int n = (int)(q / IC);
  float v = X[(size_t)n * x_n_stride + (size_t)i * HW + hw] * S[n * IC + i];
  xs[idx] = (__bf16)v;
}

// ---------------- elementwise helpers -----------------------------------
__global__ void pixel_norm_k(const float* __restrict__ x, float* __restrict__ y,
                             int rows, int cols) {
  int r = blockIdx.x; if (r >= rows) return;
  int lane = threadIdx.x;
  float s = 0.f;
  for (int c = lane; c < cols; c += 32) { float v = x[r * cols + c]; s += v * v; }
#pragma unroll
  for (int o = 16; o > 0; o >>= 1) s += __shfl_xor(s, o, 32);
  float inv = rsqrtf(s / (float)cols + 1e-8f);
  for (int c = lane; c < cols; c += 32) y[r * cols + c] = x[r * cols + c] * inv;
}

__global__ void lff_sin_k(const float* __restrict__ coords,
                          const float* __restrict__ lw,
                          const float* __restrict__ lb,
                          float* __restrict__ ff, int M, int N) {
  int idx = blockIdx.x * blockDim.x + threadIdx.x;
  if (idx >= M * N) return;
  int r = idx / N, c = idx - r * N;
  ff[idx] = sinf(coords[r] * lw[c] + lb[c]);
}

__global__ void add_comb_k(const float* __restrict__ w4,
                           const float* __restrict__ f0,
                           const float* __restrict__ f1,
                           const float* __restrict__ f2,
                           float* __restrict__ out) {
  int idx = blockIdx.x * blockDim.x + threadIdx.x;
  if (idx >= 32 * 512) return;
  int r = idx >> 9, c = idx & 511;
  out[idx] = w4[(r >> 3) * 512 + c] + f0[idx] + f1[idx] + f2[idx];
}

__global__ void w2_k(const float* __restrict__ w, float* __restrict__ W2,
                     int OC, int IC, int kk2) {
  int idx = blockIdx.x * blockDim.x + threadIdx.x;
  if (idx >= OC * IC) return;
  float s = 0.f;
  const float* p = w + (size_t)idx * kk2;
  for (int t = 0; t < kk2; ++t) s += p[t] * p[t];
  W2[idx] = s;
}

__global__ void demod_k(const float* __restrict__ s, const float* __restrict__ W2,
                        float* __restrict__ d, int BT, int OC, int IC,
                        float scale2) {
  int idx = blockIdx.x * blockDim.x + threadIdx.x;
  if (idx >= BT * OC) return;
  int n = idx / OC, o = idx - n * OC;
  float acc = 0.f;
  for (int i = 0; i < IC; ++i) {
    float sv = s[n * IC + i];
    acc += sv * sv * W2[(size_t)o * IC + i];
  }
  d[idx] = rsqrtf(scale2 * acc + 1e-8f);
}

// 4x4 blur (pad 1,1) after conv_transpose + demod/noise/act_b/lrelu epilogue
__global__ void blur_epi_k(const float* __restrict__ yT, float* __restrict__ out,
                           const float* __restrict__ dmod,
                           const float* __restrict__ noise,
                           const float* __restrict__ noiseW,
                           const float* __restrict__ actb,
                           int BT, int C, int S, int O) {
  size_t idx = (size_t)blockIdx.x * blockDim.x + threadIdx.x;
  size_t total = (size_t)BT * C * O * O;
  if (idx >= total) return;
  int u = idx % O; size_t t1 = idx / O;
  int t = t1 % O; size_t t2 = t1 / O;
  int c = t2 % C; int n = (int)(t2 / C);
  const float bl[4] = {1.f, 3.f, 3.f, 1.f};
  const float* src = yT + ((size_t)n * C + c) * S * S;
  float acc = 0.f;
  for (int a = 0; a < 4; ++a) {
    int ty = t + a - 1;
    if (ty < 0 || ty >= S) continue;
    float ra = 0.f;
    for (int b2 = 0; b2 < 4; ++b2) {
      int tx = u + b2 - 1;
      if (tx < 0 || tx >= S) continue;
      ra += bl[b2] * src[(size_t)ty * S + tx];
    }
    acc += bl[a] * ra;
  }
  acc *= (1.f / 16.f);                 // blur_kernel()*4  ==  outer/16
  acc *= dmod[n * C + c];
  acc += noiseW[0] * noise[((size_t)n * O + t) * O + u];
  acc += actb[c];
  out[idx] = (acc > 0.f ? acc : 0.2f * acc) * 1.4142135623730951f;
}

// skip-path upsample (dilate-2, 4x4 blur, pad (2,2)) accumulated into new rgb
__global__ void skip_up_add_k(const float* __restrict__ sin_,
                              float* __restrict__ sout, int BT, int H, int O) {
  size_t idx = (size_t)blockIdx.x * blockDim.x + threadIdx.x;
  size_t total = (size_t)BT * 3 * O * O;
  if (idx >= total) return;
  int u = idx % O; size_t t1 = idx / O;
  int t = t1 % O; size_t t2 = t1 / O;
  int c = t2 % 3; int n = (int)(t2 / 3);
  const float bl[4] = {1.f, 3.f, 3.f, 1.f};
  const float* src = sin_ + ((size_t)n * 3 + c) * H * H;
  float acc = 0.f;
  for (int a = 0; a < 4; ++a) {
    int uy = t + a - 2;
    if (uy < 0 || (uy & 1)) continue;
    int ih = uy >> 1; if (ih >= H) continue;
    for (int b2 = 0; b2 < 4; ++b2) {
      int ux = u + b2 - 2;
      if (ux < 0 || (ux & 1)) continue;
      int iw = ux >> 1; if (iw >= H) continue;
      acc += bl[a] * bl[b2] * src[(size_t)ih * H + iw];
    }
  }
  sout[idx] += acc * (1.f / 16.f);
}

// (BT,3,H,W) -> (B,3,T,H,W)
__global__ void permute_out_k(const float* __restrict__ skip,
                              float* __restrict__ out, int Bv, int Tv, int HW) {
  size_t idx = (size_t)blockIdx.x * blockDim.x + threadIdx.x;
  size_t total = (size_t)Bv * 3 * Tv * HW;
  if (idx >= total) return;
  int p = idx % HW; size_t q = idx / HW;
  int t = q % Tv; q /= Tv;
  int c = q % 3;  int b = (int)(q / 3);
  out[idx] = skip[(((size_t)(b * Tv + t) * 3 + c) * HW) + p];
}

// ---------------------------------------------------------------------------
// Host orchestration
// ---------------------------------------------------------------------------
static inline int cdiv(int a, int b) { return (a + b - 1) / b; }

static GemmP baseP() {
  GemmP p; memset(&p, 0, sizeof(p));
  p.a_row_div = 1; p.amul_row_div = 1;
  p.bias_scale = 1.f; p.OH = 1; p.OW = 1;
  return p;
}

template<int AMODE, int KSZ, bool AMUL, int NT>
static void g_launch(hipStream_t st, const GemmP& p) {
  int tiles = cdiv(p.M, 16) * cdiv(cdiv(p.N, 16), NT);
  hipLaunchKernelGGL((wmma_gemm<AMODE, KSZ, AMUL, NT>), dim3(cdiv(tiles, 8)),
                     dim3(256), 0, st, p);
}

extern "C" void kernel_launch(void* const* d_in, const int* in_sizes, int n_in,
                              void* d_out, int out_size, void* d_ws, size_t ws_size,
                              hipStream_t stream) {
  (void)in_sizes; (void)n_in; (void)out_size; (void)ws_size;
  // Input flattening assumption: top-level dict in insertion order
  // (z_m, z_c, coords, noises[0..10], params), params dict key-sorted
  // (const, conv1, convs, lff, motion_linear, motion_mlp, style_mlp,
  //  to_rgbs, trgb1), nested dicts key-sorted, tuples/lists in order.
  const float** in = (const float**)d_in;
  const int IDX_ZM = 0, IDX_ZC = 1, IDX_COORD = 2, IDX_NOISE = 3;
  const int IDX_CONST = 14;
  const int C1_ACTB = 15, C1_MODB = 16, C1_MODW = 17, C1_NW = 18, C1_W = 19;
  const int CONVS = 20;   // 10 x {act_b, mod_b, mod_w, noise_w, w}
  const int LFF = 70;     // 3 x {lw, lb}
  const int ML = 76;      // 3 x {eq_w, eq_b, st_act_b, st_mod_b, st_mod_w, st_w}
  const int MM = 94;      // 2 x {w, b}
  const int SM = 98;      // 8 x {w, b}
  const int TRGB = 114;   // 5 x {bias, mod_b, mod_w, w}
  const int T1_BIAS = 134, T1_MODB = 135, T1_MODW = 136, T1_W = 137;

  const float inv512 = 0.044194173824159216f; // 1/sqrt(512)

  // ---- workspace bump allocator (f32 elements, 256B aligned) ----
  float* ws = (float*)d_ws;
  size_t off = 0;
  auto alloc = [&](size_t elems) -> float* {
    float* p = ws + off; off += (elems + 63) & ~(size_t)63; return p;
  };
  float* pn_zm  = alloc(4 * 512);
  float* zm_a   = alloc(4 * 512);
  float* zm_b   = alloc(4 * 512);
  float* pn_zc  = alloc(4 * 512);
  float* w_a    = alloc(4 * 512);
  float* w_b    = alloc(4 * 512);
  float* ff_sin = alloc(32 * 512);
  float* s4     = alloc(4 * 512);
  float* ff_st  = alloc(32 * 512);
  float* ff_eq0 = alloc(32 * 512);
  float* ff_eq1 = alloc(32 * 512);
  float* ff_eq2 = alloc(32 * 512);
  float* w_comb = alloc(32 * 512);
  float* s_buf  = alloc(32 * 512);
  float* W2_buf = alloc(512 * 512);
  float* d_buf  = alloc(32 * 512);
  // packed-weight arena: max = N=512,K=4608 -> 32*144*512 bf16 elements
  __bf16* WPK = (__bf16*)alloc((size_t)32 * 144 * 512 / 2 + 64);
  // packed-activation arena: max = 32*256*128*128 bf16 elements
  __bf16* XSP = (__bf16*)alloc((size_t)32 * 256 * 128 * 128 / 2 + 64);
  float* skipA  = alloc((size_t)32 * 3 * 128 * 128);
  float* skipB  = alloc((size_t)32 * 3 * 128 * 128);
  float* P0     = alloc((size_t)32 * 256 * 128 * 128);  // act ping
  float* P1     = alloc((size_t)32 * 256 * 128 * 128);  // act pong
  float* BUF_T  = alloc((size_t)32 * 256 * 129 * 129);  // convT staging

  auto pack_w = [&](const float* W, int N, int K, float scale, int flip) {
    int Kb = K / 32;
    size_t tot = (size_t)cdiv(N, 16) * Kb * 512;
    hipLaunchKernelGGL(pack_w_k, dim3((unsigned)((tot + 255) / 256)), dim3(256),
                       0, stream, W, WPK, N, K, Kb, scale, flip);
  };
  auto pack_xs = [&](const float* X, const float* S, int IC, int HW, int xstride) {
    size_t tot = (size_t)32 * IC * HW;
    hipLaunchKernelGGL(pack_xs_k, dim3((unsigned)((tot + 255) / 256)), dim3(256),
                       0, stream, X, S, XSP, 32, IC, HW, xstride);
  };

  auto lin = [&](const float* A, int adiv, const float* Amul, int amdiv,
                 const float* B, float bscale, const float* bias, float bsc,
                 int act, float* Y, int M, int N, int K) {
    pack_w(B, N, K, bscale, 0);
    GemmP p = baseP();
    p.A = A; p.a_row_div = adiv; p.Amul = Amul; p.amul_row_div = amdiv;
    p.Bp = WPK; p.Bias = bias; p.bias_scale = bsc;
    p.act = act; p.Y = Y; p.M = M; p.N = N; p.K = K; p.Kb = K / 32;
    if (Amul) g_launch<0, 1, true, 1>(stream, p);
    else      g_launch<0, 1, false, 1>(stream, p);
  };
  auto mod_s = [&](const float* modw, const float* modb, int ICn) {
    lin(w_comb, 1, nullptr, 1, modw, inv512, modb, 1.f, 0, s_buf, 32, ICn, 512);
  };
  auto demod = [&](const float* wconv, int ICn, int OCn, int kk2) {
    int t1 = OCn * ICn;
    hipLaunchKernelGGL(w2_k, dim3(cdiv(t1, 256)), dim3(256), 0, stream,
                       wconv, W2_buf, OCn, ICn, kk2);
    int t2 = 32 * OCn;
    hipLaunchKernelGGL(demod_k, dim3(cdiv(t2, 256)), dim3(256), 0, stream,
                       s_buf, W2_buf, d_buf, 32, OCn, ICn,
                       1.f / (float)(ICn * kk2));
  };

  // ---- motion MLP (lr_mul 0.01): pixel_norm + 2 equal_linear(lrelu) ----
  hipLaunchKernelGGL(pixel_norm_k, dim3(4), dim3(32), 0, stream, in[IDX_ZM], pn_zm, 4, 512);
  lin(pn_zm, 1, nullptr, 1, in[MM + 0], 0.01f * inv512, in[MM + 1], 0.01f, 1, zm_a, 4, 512, 512);
  lin(zm_a,  1, nullptr, 1, in[MM + 2], 0.01f * inv512, in[MM + 3], 0.01f, 1, zm_b, 4, 512, 512);

  // ---- style MLP (8 layers, lr_mul 0.01) ----
  hipLaunchKernelGGL(pixel_norm_k, dim3(4), dim3(32), 0, stream, in[IDX_ZC], pn_zc, 4, 512);
  const float* cur = pn_zc;
  float* wbufs[2] = {w_a, w_b};
  for (int l = 0; l < 8; ++l) {
    lin(cur, 1, nullptr, 1, in[SM + 2 * l], 0.01f * inv512, in[SM + 2 * l + 1],
        0.01f, 1, wbufs[l & 1], 4, 512, 512);
    cur = wbufs[l & 1];
  }
  const float* w_final = cur;                 // (4,512), frame rows repeat x8

  // ---- LFF fourier features x3: sin -> styled_linear -> equal_linear ----
  float* ff_eq[3] = {ff_eq0, ff_eq1, ff_eq2};
  for (int f = 0; f < 3; ++f) {
    hipLaunchKernelGGL(lff_sin_k, dim3(cdiv(32 * 512, 256)), dim3(256), 0, stream,
                       in[IDX_COORD], in[LFF + 2 * f], in[LFF + 2 * f + 1],
                       ff_sin, 32, 512);
    // s = zm_rep @ (mod_w/sqrt512).T + mod_b   (rows repeat -> compute M=4)
    lin(zm_b, 1, nullptr, 1, in[ML + 6 * f + 4], inv512, in[ML + 6 * f + 3],
        1.f, 0, s4, 4, 512, 512);
    // lrelu((ff*s) @ (w/sqrt512).T + act_b)
    lin(ff_sin, 1, s4, 8, in[ML + 6 * f + 5], inv512, in[ML + 6 * f + 2],
        1.f, 1, ff_st, 32, 512, 512);
    // equal_linear no-act
    lin(ff_st, 1, nullptr, 1, in[ML + 6 * f + 0], inv512, in[ML + 6 * f + 1],
        1.f, 0, ff_eq[f], 32, 512, 512);
  }
  hipLaunchKernelGGL(add_comb_k, dim3(cdiv(32 * 512, 256)), dim3(256), 0, stream,
                     w_final, ff_eq0, ff_eq1, ff_eq2, w_comb);
  // NOTE: latent = repeat(w_comb, 12) -> every layer uses the same w_comb.

  // ---- conv1: styled 3x3 @ 4x4, 512->512, input = tiled const ----
  mod_s(in[C1_MODW], in[C1_MODB], 512);
  demod(in[C1_W], 512, 512, 9);
  pack_xs(in[IDX_CONST], s_buf, 512, 16, 0);
  pack_w(in[C1_W], 512, 4608, 1.f / sqrtf(4608.f), 0);
  {
    GemmP p = baseP();
    p.Xs = XSP; p.Bp = WPK;
    p.IC = 512; p.H = 4; p.W = 4; p.OH = 4; p.OW = 4; p.pad = 1;
    p.M = 32 * 16; p.N = 512; p.K = 4608; p.Kb = 144;
    p.Dmod = d_buf; p.Noise = in[IDX_NOISE + 0]; p.NoiseW = in[C1_NW];
    p.Bias = in[C1_ACTB]; p.act = 1; p.out_mode = 1; p.Y = P0;
    g_launch<1, 3, false, 4>(stream, p);
  }
  // ---- trgb1: 1x1 modulated conv (no demod) + bias -> skipA ----
  mod_s(in[T1_MODW], in[T1_MODB], 512);
  pack_xs(P0, s_buf, 512, 16, 512 * 16);
  pack_w(in[T1_W], 3, 512, inv512, 0);
  {
    GemmP p = baseP();
    p.Xs = XSP; p.Bp = WPK;
    p.IC = 512; p.H = 4; p.W = 4; p.OH = 4; p.OW = 4; p.pad = 0;
    p.M = 32 * 16; p.N = 3; p.K = 512; p.Kb = 16;
    p.Bias = in[T1_BIAS]; p.out_mode = 1; p.Y = skipA;
    g_launch<1, 1, false, 1>(stream, p);
  }

  // ---- synthesis stages: 4->8->16->32->64->128 ----
  float* skip_cur = skipA; float* skip_nxt = skipB;
  int Hc = 4, ICc = 512;
  for (int li = 0; li < 5; ++li) {
    const int OC = (li == 4) ? 256 : 512;
    const int O = 2 * Hc, S = 2 * Hc + 1;
    const int cu = CONVS + 5 * (2 * li);      // upsample conv leaves
    const int cr = CONVS + 5 * (2 * li + 1);  // regular conv leaves
    const int tb = TRGB + 4 * li;

    // upsample styled conv: convT GEMM -> blur(+demod/noise/act) epilogue
    mod_s(in[cu + 2], in[cu + 1], ICc);
    demod(in[cu + 4], ICc, OC, 9);
    pack_xs(P0, s_buf, ICc, Hc * Hc, ICc * Hc * Hc);
    pack_w(in[cu + 4], OC, ICc * 9, 1.f / sqrtf((float)(ICc * 9)), 1);
    {
      GemmP p = baseP();
      p.Xs = XSP; p.Bp = WPK;
      p.IC = ICc; p.H = Hc; p.W = Hc; p.OH = S; p.OW = S;
      p.M = 32 * S * S; p.N = OC; p.K = ICc * 9; p.Kb = ICc * 9 / 32;
      p.out_mode = 1; p.Y = BUF_T;
      g_launch<2, 3, false, 4>(stream, p);
    }
    {
      size_t tot = (size_t)32 * OC * O * O;
      hipLaunchKernelGGL(blur_epi_k, dim3((unsigned)((tot + 255) / 256)), dim3(256),
                         0, stream, BUF_T, P1, d_buf, in[IDX_NOISE + 1 + 2 * li],
                         in[cu + 3], in[cu + 0], 32, OC, S, O);
    }
    // regular styled 3x3 conv at O
    mod_s(in[cr + 2], in[cr + 1], OC);
    demod(in[cr + 4], OC, OC, 9);
    pack_xs(P1, s_buf, OC, O * O, OC * O * O);
    pack_w(in[cr + 4], OC, OC * 9, 1.f / sqrtf((float)(OC * 9)), 0);
    {
      GemmP p = baseP();
      p.Xs = XSP; p.Bp = WPK;
      p.IC = OC; p.H = O; p.W = O; p.OH = O; p.OW = O; p.pad = 1;
      p.M = 32 * O * O; p.N = OC; p.K = OC * 9; p.Kb = OC * 9 / 32;
      p.Dmod = d_buf; p.Noise = in[IDX_NOISE + 2 + 2 * li]; p.NoiseW = in[cr + 3];
      p.Bias = in[cr + 0]; p.act = 1; p.out_mode = 1; p.Y = P0;
      g_launch<1, 3, false, 4>(stream, p);
    }
    // to_rgb (1x1, no demod) + upsampled skip accumulation
    mod_s(in[tb + 2], in[tb + 1], OC);
    pack_xs(P0, s_buf, OC, O * O, OC * O * O);
    pack_w(in[tb + 3], 3, OC, 1.f / sqrtf((float)OC), 0);
    {
      GemmP p = baseP();
      p.Xs = XSP; p.Bp = WPK;
      p.IC = OC; p.H = O; p.W = O; p.OH = O; p.OW = O; p.pad = 0;
      p.M = 32 * O * O; p.N = 3; p.K = OC; p.Kb = OC / 32;
      p.Bias = in[tb + 0]; p.out_mode = 1; p.Y = skip_nxt;
      g_launch<1, 1, false, 1>(stream, p);
    }
    {
      size_t tot = (size_t)32 * 3 * O * O;
      hipLaunchKernelGGL(skip_up_add_k, dim3((unsigned)((tot + 255) / 256)), dim3(256),
                         0, stream, skip_cur, skip_nxt, 32, Hc, O);
    }
    float* tmp = skip_cur; skip_cur = skip_nxt; skip_nxt = tmp;
    Hc = O; ICc = OC;
  }

  // ---- (BT,3,128,128) -> (B,3,T,128,128) ----
  hipLaunchKernelGGL(permute_out_k, dim3(cdiv(4 * 3 * 8 * 128 * 128, 256)), dim3(256),
                     0, stream, skip_cur, (float*)d_out, 4, 8, 128 * 128);
}
